// Attention_36318243455565
// MI455X (gfx1250) — matrix operand order repaired and checked by reading, not compiled, not run on hardware
//
#include <hip/hip_runtime.h>
#include <hip/hip_bf16.h>

// ---------------------------------------------------------------------------
// Causal self-attention for MI455X (gfx1250, wave32, WMMA).
//   Kernel 1: Q/K/V = x @ W^T + b  (fp32 in, bf16 out to workspace)
//   Kernel 2: flash attention (bf16 WMMA, fp32 accum, online softmax, relu)
//             K tiles staged with GLOBAL_LOAD_ASYNC_TO_LDS_B128 (ASYNCcnt)
// Workspace: 3 * 32*1024*512 bf16 = ~100.7 MB.
// ---------------------------------------------------------------------------

typedef __bf16 bf16;
typedef __attribute__((ext_vector_type(16))) __bf16 v16bf;
typedef __attribute__((ext_vector_type(8)))  __bf16 v8bf;
typedef __attribute__((ext_vector_type(8)))  float  v8f;
typedef __attribute__((ext_vector_type(4)))  float  v4f;

#define BATCH 32
#define SEQ   1024
#define DMODEL 1024
#define DHEAD  512

static __device__ __forceinline__ v16bf cat16(v8bf lo, v8bf hi) {
    return __builtin_shufflevector(lo, hi, 0,1,2,3,4,5,6,7,8,9,10,11,12,13,14,15);
}
static __device__ __forceinline__ v8f wmma_bf16(v16bf a, v16bf b, v8f c) {
    // D = A(16x32) * B(32x16) + C, fp32 accumulate
    return __builtin_amdgcn_wmma_f32_16x16x32_bf16(false, a, false, b, (short)0, c, false, false);
}

// ---------------------------------------------------------------------------
// Kernel 1: QKV projection. Grid (256, 8, 3), block 256 (8 waves).
// Block computes a 128x64 tile of Y = X * W^T + b for matrix z (0=q,1=k,2=v).
// ---------------------------------------------------------------------------
__global__ __launch_bounds__(256) void qkv_proj_kernel(
    const float* __restrict__ x,
    const float* __restrict__ Wq, const float* __restrict__ bq,
    const float* __restrict__ Wk, const float* __restrict__ bk,
    const float* __restrict__ Wv, const float* __restrict__ bv,
    bf16* __restrict__ yqkv)
{
    const int tid   = threadIdx.x;
    const int m0    = blockIdx.x * 128;          // row tile over B*L = 32768
    const int n0    = blockIdx.y * 64;           // col tile over D = 512
    const int which = blockIdx.z;                // 0=q, 1=k, 2=v
    const float* W    = (which == 0) ? Wq : (which == 1) ? Wk : Wv;
    const float* bias = (which == 0) ? bq : (which == 1) ? bk : bv;
    bf16* Y = yqkv + (size_t)which * ((size_t)BATCH * SEQ * DHEAD);

    // +8 bf16 pad keeps row stride a multiple of 16B and spreads LDS banks.
    __shared__ alignas(16) bf16 sX[128][40];
    __shared__ alignas(16) bf16 sW[64][40];

    const int wv = tid >> 5, lane = tid & 31;
    const int lm = lane & 15, lh = lane >> 4;
    const int cg = (tid & 7) * 4;   // column group (4 floats)
    const int r0 = tid >> 3;        // 0..31

    v8f acc[4] = {};

    for (int kc = 0; kc < DMODEL; kc += 32) {
        // Stage X tile (128x32 fp32 -> bf16) and W tile (64x32) into LDS.
        #pragma unroll
        for (int i = 0; i < 4; ++i) {
            int row = r0 + 32 * i;
            v4f xv = *(const v4f*)&x[(size_t)(m0 + row) * DMODEL + kc + cg];
            #pragma unroll
            for (int j = 0; j < 4; ++j) sX[row][cg + j] = (bf16)xv[j];
        }
        #pragma unroll
        for (int i = 0; i < 2; ++i) {
            int row = r0 + 32 * i;
            v4f wvv = *(const v4f*)&W[(size_t)(n0 + row) * DMODEL + kc + cg];
            #pragma unroll
            for (int j = 0; j < 4; ++j) sW[row][cg + j] = (bf16)wvv[j];
        }
        __syncthreads();

        // A fragment: lane holds row (16w+lm), K = {8h..8h+7} U {16+8h..16+8h+7}
        v16bf a = cat16(*(const v8bf*)&sX[wv * 16 + lm][8 * lh],
                        *(const v8bf*)&sX[wv * 16 + lm][16 + 8 * lh]);
        #pragma unroll
        for (int t = 0; t < 4; ++t) {
            // B fragment: lane holds col (16t+lm), K = {8h..8h+7} U {16+8h..16+8h+7}
            v16bf bm = cat16(*(const v8bf*)&sW[t * 16 + lm][8 * lh],
                             *(const v8bf*)&sW[t * 16 + lm][16 + 8 * lh]);
            acc[t] = wmma_bf16(a, bm, acc[t]);
        }
        __syncthreads();
    }

    // Epilogue: bias add, convert to bf16, store. C layout: row = g+8h, col = lm.
    #pragma unroll
    for (int t = 0; t < 4; ++t) {
        float bb = bias[n0 + t * 16 + lm];
        #pragma unroll
        for (int g = 0; g < 8; ++g) {
            int row = m0 + wv * 16 + g + 8 * lh;
            int col = n0 + t * 16 + lm;
            Y[(size_t)row * DHEAD + col] = (bf16)(acc[t][g] + bb);
        }
    }
}

// ---------------------------------------------------------------------------
// Kernel 2: causal flash attention + relu.
// Grid (32, 16, 2), block 128 (4 waves). Block handles (batch b, 64 query
// rows, 256 output columns). Each wave: 16 query rows, full-D QK^T (bf16
// WMMA), online softmax in fp32, PV over its 256-col half.
// ---------------------------------------------------------------------------
__global__ __launch_bounds__(128) void attn_kernel(
    const bf16* __restrict__ qkv,
    const int*  __restrict__ mask,
    float* __restrict__ out)
{
    const int b  = blockIdx.x;
    const int qt = blockIdx.y;
    const int dh = blockIdx.z;
    const int tid = threadIdx.x, wv = tid >> 5, lane = tid & 31;
    const int lm = lane & 15, lh = lane >> 4;
    const int qbase = qt * 64;
    const int d0 = dh * 256;

    const size_t MS = (size_t)BATCH * SEQ * DHEAD;
    const bf16* Q = qkv;
    const bf16* K = qkv + MS;
    const bf16* V = qkv + 2 * MS;

    __shared__ alignas(16) bf16 sK[32][520];      // 32 keys x 512 d (+pad)
    __shared__ alignas(16) bf16 sVt[256][40];     // transposed V half: d x key
    __shared__ alignas(16) bf16 sP[4][16][40];    // per-wave P bounce buffer

    v8f o[16] = {};                               // 16 rows x 256 cols fp32
    float mrow[8], lrow[8];
    #pragma unroll
    for (int g = 0; g < 8; ++g) { mrow[g] = -1e30f; lrow[g] = 0.0f; }

    const bf16* qp = Q + (size_t)(b * SEQ + qbase + wv * 16 + lm) * DHEAD;
    const float scale = 0.044194173824159216f;    // 512^-0.5

    const int nchunk = (qbase + 64) >> 5;         // causal: keys <= qbase+63

    for (int kc = 0; kc < nchunk; ++kc) {
        const int kbase = kc * 32;

        // Prefetch next chunk's K rows into cache (global_prefetch_b8).
        if (kc + 1 < nchunk)
            __builtin_prefetch(&K[(size_t)(b * SEQ + kbase + 32 + (tid & 31)) * DHEAD
                                  + (tid >> 5) * 128], 0, 1);

        // ---- K tile: async global -> LDS DMA (no VGPR round trip). --------
        // Each lane copies 16B; 128 threads x 16 iters = 32 KB (32 x 512 bf16).
        // Lane's LDS dest = low 32 bits of the generic shared-pointer (LDS
        // aperture truncation, ISA 10.2). Tracked by ASYNCcnt.
        #pragma unroll
        for (int i = 0; i < 16; ++i) {
            int row = (tid >> 6) + 2 * i;          // 0..31
            int c8  = (tid & 63) * 8;              // 0..504
            unsigned ldsoff = (unsigned)(size_t)&sK[row][c8];
            const bf16* gsrc = &K[(size_t)(b * SEQ + kbase + row) * DHEAD + c8];
            asm volatile("global_load_async_to_lds_b128 %0, %1, off"
                         :: "v"(ldsoff), "v"(gsrc) : "memory");
        }
        // ---- V-half staged transposed through VGPRs (needs shuffle). ------
        {
            int key = tid & 31, dg = tid >> 5;     // dg 0..3 -> 64 d each
            #pragma unroll
            for (int i = 0; i < 8; ++i) {
                int dl = dg * 64 + i * 8;
                v8bf vvv = *(const v8bf*)&V[(size_t)(b * SEQ + kbase + key) * DHEAD + d0 + dl];
                #pragma unroll
                for (int j = 0; j < 8; ++j) sVt[dl + j][key] = vvv[j];
            }
        }
        asm volatile("s_wait_asynccnt 0" ::: "memory");  // K DMA landed in LDS
        __syncthreads();

        // ---- S = Q K^T over full D=512 (two 16x16 tiles of 32 keys) ----
        v8f s0 = {}, s1 = {};
        {
            const bf16* qpc = qp;
            asm volatile("" : "+v"(qpc));          // keep Q loads inside loop (L2 hits)
            #pragma unroll
            for (int c = 0; c < 16; ++c) {
                v16bf a = cat16(*(const v8bf*)(qpc + 32 * c + 8 * lh),
                                *(const v8bf*)(qpc + 32 * c + 16 + 8 * lh));
                v16bf b0 = cat16(*(const v8bf*)&sK[lm][32 * c + 8 * lh],
                                 *(const v8bf*)&sK[lm][32 * c + 16 + 8 * lh]);
                v16bf b1 = cat16(*(const v8bf*)&sK[16 + lm][32 * c + 8 * lh],
                                 *(const v8bf*)&sK[16 + lm][32 * c + 16 + 8 * lh]);
                s0 = wmma_bf16(a, b0, s0);
                s1 = wmma_bf16(a, b1, s1);
            }
        }

        // ---- causal + padding mask, online softmax (rows split at lane 16) ----
        const int key0 = kbase + lm, key1 = kbase + 16 + lm;
        const bool ok0 = (mask[b * SEQ + key0] != 0);
        const bool ok1 = (mask[b * SEQ + key1] != 0);
        float alpha[8];
        #pragma unroll
        for (int g = 0; g < 8; ++g) {
            int row = qbase + wv * 16 + g + 8 * lh;
            float x0 = (ok0 && key0 <= row) ? s0[g] * scale : -1e30f;
            float x1 = (ok1 && key1 <= row) ? s1[g] * scale : -1e30f;
            float mx = fmaxf(x0, x1);
            mx = fmaxf(mx, __shfl_xor(mx, 1));
            mx = fmaxf(mx, __shfl_xor(mx, 2));
            mx = fmaxf(mx, __shfl_xor(mx, 4));
            mx = fmaxf(mx, __shfl_xor(mx, 8));
            float mnew = fmaxf(mrow[g], mx);
            float a  = __expf(mrow[g] - mnew);
            float p0 = __expf(x0 - mnew);
            float p1 = __expf(x1 - mnew);
            float rs = p0 + p1;
            rs += __shfl_xor(rs, 1);
            rs += __shfl_xor(rs, 2);
            rs += __shfl_xor(rs, 4);
            rs += __shfl_xor(rs, 8);
            lrow[g] = lrow[g] * a + rs;
            mrow[g] = mnew;
            alpha[g] = a;
            s0[g] = p0; s1[g] = p1;
        }

        // Rescale O accumulators by alpha (C layout row matches g+8h).
        #pragma unroll
        for (int t = 0; t < 16; ++t)
            #pragma unroll
            for (int g = 0; g < 8; ++g) o[t][g] *= alpha[g];

        // ---- P (C layout) -> LDS -> A fragment (16x32 bf16) ----
        #pragma unroll
        for (int g = 0; g < 8; ++g) {
            sP[wv][g + 8 * lh][lm]      = (bf16)s0[g];
            sP[wv][g + 8 * lh][16 + lm] = (bf16)s1[g];
        }
        asm volatile("s_wait_dscnt 0" ::: "memory");  // CDNA5 split DS counter
        v16bf pa = cat16(*(const v8bf*)&sP[wv][lm][8 * lh],
                         *(const v8bf*)&sP[wv][lm][16 + 8 * lh]);

        // ---- O += P * V (16 N-subtiles of the 256-col half) ----
        #pragma unroll
        for (int t = 0; t < 16; ++t) {
            v16bf bm = cat16(*(const v8bf*)&sVt[16 * t + lm][8 * lh],
                             *(const v8bf*)&sVt[16 * t + lm][16 + 8 * lh]);
            o[t] = wmma_bf16(pa, bm, o[t]);
        }
        __syncthreads();
    }

    // ---- epilogue: normalize, relu, store fp32 ----
    #pragma unroll
    for (int t = 0; t < 16; ++t) {
        #pragma unroll
        for (int g = 0; g < 8; ++g) {
            int row = qbase + wv * 16 + g + 8 * lh;
            int col = d0 + t * 16 + lm;
            float val = o[t][g] / lrow[g];
            out[(size_t)(b * SEQ + row) * DHEAD + col] = fmaxf(val, 0.0f);
        }
    }
}

// ---------------------------------------------------------------------------
// Host launcher
// ---------------------------------------------------------------------------
extern "C" void kernel_launch(void* const* d_in, const int* in_sizes, int n_in,
                              void* d_out, int out_size, void* d_ws, size_t ws_size,
                              hipStream_t stream) {
    (void)in_sizes; (void)n_in; (void)out_size; (void)ws_size;
    const float* x  = (const float*)d_in[0];
    const float* Wq = (const float*)d_in[1];
    const float* bq = (const float*)d_in[2];
    const float* Wk = (const float*)d_in[3];
    const float* bk = (const float*)d_in[4];
    const float* Wv = (const float*)d_in[5];
    const float* bv = (const float*)d_in[6];
    const int*   mask = (const int*)d_in[7];
    float* out = (float*)d_out;
    bf16*  qkv = (bf16*)d_ws;   // 3 * 32*1024*512 bf16 = ~100.7 MB scratch

    dim3 g1((BATCH * SEQ) / 128, DHEAD / 64, 3);
    qkv_proj_kernel<<<g1, 256, 0, stream>>>(x, Wq, bq, Wk, bk, Wv, bv, qkv);

    dim3 g2(BATCH, SEQ / 64, 2);
    attn_kernel<<<g2, 128, 0, stream>>>(qkv, mask, out);
}
